// TemporalGraphAttentionNetwork_34033320854151
// MI455X (gfx1250) — matrix operand order repaired and checked by reading, not compiled, not run on hardware
//
#include <hip/hip_runtime.h>
#include <math.h>

typedef _Float16 v16h __attribute__((ext_vector_type(16)));
typedef _Float16 v8h  __attribute__((ext_vector_type(8)));
typedef float    v8f  __attribute__((ext_vector_type(8)));

#define NODES_PER_WG 8
#define NTHREADS     256
#define N_NEI        20
#define ND           128
#define TD           64
#define CD           192   // ND + TD
#define HD           128
#define NH           4
#define DH           32
#define NLAYERS      2
#define ROWS   (NODES_PER_WG * N_NEI)   // 160
#define MTILES (ROWS / 16)              // 10

// ---- f16 transposed-weight workspace layout (element offsets, _Float16) ----
#define OFF_WQ  0
#define SZ_QKV  (NLAYERS * HD * CD)          // 49152 each
#define OFF_WK  (OFF_WQ + SZ_QKV)
#define OFF_WV  (OFF_WK + SZ_QKV)
#define OFF_WO  (OFF_WV + SZ_QKV)
#define SZ_WO   (NLAYERS * ND * HD)          // 32768
#define OFF_W1  (OFF_WO + SZ_WO)
#define SZ_W1   (NLAYERS * 2 * ND * ND)      // 65536
#define OFF_W2  (OFF_W1 + SZ_W1)
#define SZ_W2   (NLAYERS * ND * 2 * ND)      // 65536
#define OFF_WC1 (OFF_W2 + SZ_W2)
#define SZ_WC1  ((ND / 2) * ND)              // 8192
#define WS_HALVES (OFF_WC1 + SZ_WC1)         // 319488 (~624 KB)

// ---------------- weight prep: f32 row-major [K][N] -> f16 [N][K] ----------------
__global__ __launch_bounds__(256)
void prep_weights(const float* __restrict__ Wq, const float* __restrict__ Wk,
                  const float* __restrict__ Wv, const float* __restrict__ Wo,
                  const float* __restrict__ W1, const float* __restrict__ W2,
                  const float* __restrict__ Wc1, _Float16* __restrict__ ws) {
  const int i0 = blockIdx.x * blockDim.x + threadIdx.x;
  const int st = gridDim.x * blockDim.x;
  // Wq/Wk/Wv: src [l][CD][HD] -> dst [l][HD][CD]
  for (int t = i0; t < NLAYERS * HD * CD; t += st) {
    int l = t / (HD * CD), r = t % (HD * CD);
    int n = r / CD, k = r % CD;
    size_t src = (size_t)l * CD * HD + (size_t)k * HD + n;
    ws[OFF_WQ + t] = (_Float16)Wq[src];
    ws[OFF_WK + t] = (_Float16)Wk[src];
    ws[OFF_WV + t] = (_Float16)Wv[src];
  }
  // Wo: src [l][HD][ND] -> dst [l][ND][HD]
  for (int t = i0; t < NLAYERS * ND * HD; t += st) {
    int l = t / (ND * HD), r = t % (ND * HD);
    int n = r / HD, k = r % HD;
    ws[OFF_WO + t] = (_Float16)Wo[(size_t)l * HD * ND + (size_t)k * ND + n];
  }
  // W1: src [l][ND][2ND] -> dst [l][2ND][ND]
  for (int t = i0; t < NLAYERS * 2 * ND * ND; t += st) {
    int l = t / (2 * ND * ND), r = t % (2 * ND * ND);
    int n = r / ND, k = r % ND;
    ws[OFF_W1 + t] = (_Float16)W1[(size_t)l * ND * 2 * ND + (size_t)k * 2 * ND + n];
  }
  // W2: src [l][2ND][ND] -> dst [l][ND][2ND]
  for (int t = i0; t < NLAYERS * ND * 2 * ND; t += st) {
    int l = t / (ND * 2 * ND), r = t % (ND * 2 * ND);
    int n = r / (2 * ND), k = r % (2 * ND);
    ws[OFF_W2 + t] = (_Float16)W2[(size_t)l * 2 * ND * ND + (size_t)k * ND + n];
  }
  // Wc1: src [ND][ND/2] -> dst [ND/2][ND]
  for (int t = i0; t < (ND / 2) * ND; t += st) {
    int n = t / ND, k = t % ND;
    ws[OFF_WC1 + t] = (_Float16)Wc1[(size_t)k * (ND / 2) + n];
  }
}

// ---------------- WMMA fragment helpers (CDNA5 16x16x32 f16) ----------------

// A-matrix 16x32 f16: lanes 0-15 hold row M=lane, halves 0..15 = K0..15;
// lanes 16-31 hold K16..31.
__device__ inline v16h load_a_frag_lds(const _Float16* base, int stride, int k0, int lane) {
  const int r  = lane & 15;
  const int hs = (lane >> 4) << 4;      // 0 or 16
  const _Float16* p = base + r * stride + k0;
  v8h lo = *(const v8h*)(p + hs);
  v8h hi = *(const v8h*)(p + 8 + hs);
  return __builtin_shufflevector(lo, hi, 0,1,2,3,4,5,6,7,8,9,10,11,12,13,14,15);
}

// B-matrix 32x16 f16 from transposed f16 weights Wt[N][K]: lane n<16 holds
// column col0+n, halves 0..15 = K k0..k0+15; lanes 16-31 hold K k0+16..k0+31.
// 16 contiguous halves per lane -> two global_load_b128.
__device__ inline v16h load_b_frag_t(const _Float16* __restrict__ Wt, int K,
                                     int k0, int col0, int lane) {
  const _Float16* p = Wt + (size_t)(col0 + (lane & 15)) * K + k0 + ((lane >> 4) << 4);
  v8h lo = *(const v8h*)p;
  v8h hi = *(const v8h*)(p + 8);
  return __builtin_shufflevector(lo, hi, 0,1,2,3,4,5,6,7,8,9,10,11,12,13,14,15);
}

__device__ inline v8f wmma_f16(v16h a, v16h b, v8f c) {
  return __builtin_amdgcn_wmma_f32_16x16x32_f16(false, a, false, b, (short)0, c,
                                                false, false);
}

// residual add + layernorm over ND columns for NODES_PER_WG rows
__device__ inline void residual_layernorm(float x[][ND], float delta[][2 * ND],
                                          const float* g, const float* be,
                                          float red[][32], int tid) {
  for (int i = tid; i < NODES_PER_WG * ND; i += NTHREADS) {
    int b = i >> 7, c = i & 127;
    delta[b][c] += x[b][c];
  }
  __syncthreads();
  if (tid < NODES_PER_WG * 16) {
    int b = tid >> 4, j = tid & 15;
    float s = 0.f, ss = 0.f;
#pragma unroll
    for (int e = 0; e < 8; ++e) { float y = delta[b][j * 8 + e]; s += y; ss += y * y; }
    red[b][j] = s; red[b][16 + j] = ss;
  }
  __syncthreads();
  if (tid < NODES_PER_WG) {
    int b = tid;
    float s = 0.f, ss = 0.f;
#pragma unroll
    for (int j = 0; j < 16; ++j) { s += red[b][j]; ss += red[b][16 + j]; }
    float mean = s * (1.f / ND);
    float var  = ss * (1.f / ND) - mean * mean;
    red[b][0] = mean;
    red[b][1] = rsqrtf(var + 1e-5f);
  }
  __syncthreads();
  for (int i = tid; i < NODES_PER_WG * ND; i += NTHREADS) {
    int b = i >> 7, c = i & 127;
    float y = delta[b][c];
    x[b][c] = (y - red[b][0]) * red[b][1] * g[c] + be[c];
  }
  __syncthreads();
}

__global__ __launch_bounds__(NTHREADS)
void tgat_fused(const float* __restrict__ node_features,
                const float* __restrict__ neighbor_features,
                const float* __restrict__ node_times,
                const float* __restrict__ neighbor_times,
                const float* __restrict__ time_freq,
                const float* __restrict__ time_phase,
                const _Float16* __restrict__ wsh,   // f16 transposed weights
                const float* __restrict__ bq, const float* __restrict__ bk,
                const float* __restrict__ bv, const float* __restrict__ bo,
                const float* __restrict__ b1, const float* __restrict__ b2,
                const float* __restrict__ g1, const float* __restrict__ be1,
                const float* __restrict__ g2, const float* __restrict__ be2,
                const float* __restrict__ bc1, const float* __restrict__ Wc2,
                const float* __restrict__ bc2,
                const int* __restrict__ mask,
                float* __restrict__ out, int B) {
  __shared__ _Float16 sh_kcomb[ROWS][CD];          // 61440 B, reused both layers
  __shared__ _Float16 sh_kT[HD][ROWS];             // 40960 B, K^T (col-major)
  __shared__ float    sh_x[NODES_PER_WG][ND];      //  4096 B
  __shared__ float    sh_tq[NODES_PER_WG][TD];     //  2048 B
  __shared__ _Float16 sh_a16[16][CD];              //  6144 B  A-tile staging
  __shared__ _Float16 sh_h16[16][2 * ND];          //  8192 B  FFN hidden (f16)
  __shared__ float    sh_q[16][HD];                //  8192 B
  __shared__ float    sh_c[16][2 * ND];            // 16384 B  GEMM C staging
  __shared__ float    sh_w[NODES_PER_WG][NH][N_NEI]; // 2560 B softmax weights
  __shared__ float    sh_mb[NODES_PER_WG][N_NEI];  //   640 B mask bias
  __shared__ float    sh_attn[NODES_PER_WG][ND];   //  4096 B
  __shared__ float    sh_red[NODES_PER_WG][32];    //  1024 B

  const int tid  = threadIdx.x;
  const int lane = tid & 31;
  const int wave = tid >> 5;
  const int b0   = blockIdx.x * NODES_PER_WG;
  const float scale = 0.17677669529663687f;   // 1/sqrt(32)

  // ---------------- Phase 0: stage activations / time encodings ----------------
  // Per-node streams are touched exactly once across the whole grid -> NT loads
  // keep the 192MB L2 free for the f16 weight working set.
  for (int i = tid; i < NODES_PER_WG * ND; i += NTHREADS) {
    int b = i >> 7, c = i & 127;
    int gb = (b0 + b < B) ? b0 + b : B - 1;
    sh_x[b][c] = __builtin_nontemporal_load(&node_features[(size_t)gb * ND + c]);
  }
  for (int i = tid; i < NODES_PER_WG * TD; i += NTHREADS) {
    int b = i / TD, c = i % TD;
    int gb = (b0 + b < B) ? b0 + b : B - 1;
    sh_tq[b][c] = cosf(__builtin_nontemporal_load(&node_times[gb]) * time_freq[c] +
                       time_phase[c]);
  }
  for (int i = tid; i < NODES_PER_WG * N_NEI; i += NTHREADS) {
    int b = i / N_NEI, n = i % N_NEI;
    int gb = (b0 + b < B) ? b0 + b : B - 1;
    int m = __builtin_nontemporal_load(&mask[(size_t)gb * N_NEI + n]);
    sh_mb[b][n] = (m == 0) ? -1e9f : 0.0f;
  }
  for (int i = tid; i < ROWS * CD; i += NTHREADS) {
    int r = i / CD, c = i % CD;
    int b = r / N_NEI, n = r % N_NEI;
    int gb = (b0 + b < B) ? b0 + b : B - 1;
    float val;
    if (c < ND) {
      val = __builtin_nontemporal_load(
          &neighbor_features[((size_t)gb * N_NEI + n) * ND + c]);
    } else {
      int t = c - ND;
      float tt = __builtin_nontemporal_load(&neighbor_times[(size_t)gb * N_NEI + n]);
      val = cosf(tt * time_freq[t] + time_phase[t]);
    }
    sh_kcomb[r][c] = (_Float16)val;
  }
  __syncthreads();

  // ---------------- Transformer layers ----------------
  for (int l = 0; l < NLAYERS; ++l) {
    const _Float16* WqT = wsh + OFF_WQ + (size_t)l * HD * CD;
    const _Float16* WkT = wsh + OFF_WK + (size_t)l * HD * CD;
    const _Float16* WvT = wsh + OFF_WV + (size_t)l * HD * CD;
    const _Float16* WoT = wsh + OFF_WO + (size_t)l * ND * HD;
    const _Float16* W1T = wsh + OFF_W1 + (size_t)l * 2 * ND * ND;
    const _Float16* W2T = wsh + OFF_W2 + (size_t)l * ND * 2 * ND;

    // (a) qcomb A-tile (rows >= 8 zero)
    for (int i = tid; i < 16 * CD; i += NTHREADS) {
      int r = i / CD, c = i % CD;
      float v = 0.f;
      if (r < NODES_PER_WG) v = (c < ND) ? sh_x[r][c] : sh_tq[r][c - ND];
      sh_a16[r][c] = (_Float16)v;
    }
    __syncthreads();

    // (b) q = qcomb @ Wq + bq  (one M-tile, wave w owns columns 16w..16w+15)
    {
      const int col0 = wave * 16;
      const int colg = col0 + (lane & 15);
      v8f c = {};
#pragma unroll
      for (int kf = 0; kf < 6; ++kf) {
        v16h a = load_a_frag_lds(&sh_a16[0][0], CD, 32 * kf, lane);
        v16h bf = load_b_frag_t(WqT, CD, 32 * kf, col0, lane);
        c = wmma_f16(a, bf, c);
      }
      const float bias = bq[l * HD + colg];
      const int r0 = (lane >> 4) * 8;
#pragma unroll
      for (int r = 0; r < 8; ++r) sh_q[r0 + r][colg] = c[r] + bias;
    }

    // (c) K = kcomb @ Wk + bk -> sh_kT (f16, col-major); B frags hoisted
    {
      const int col0 = wave * 16;
      const int colg = col0 + (lane & 15);
      const float biask = bk[l * HD + colg];
      v16h bfrag[6];
#pragma unroll
      for (int kf = 0; kf < 6; ++kf)
        bfrag[kf] = load_b_frag_t(WkT, CD, 32 * kf, col0, lane);
      for (int t = 0; t < MTILES; ++t) {
        v8f c = {};
#pragma unroll
        for (int kf = 0; kf < 6; ++kf) {
          v16h a = load_a_frag_lds(&sh_kcomb[t * 16][0], CD, 32 * kf, lane);
          c = wmma_f16(a, bfrag[kf], c);
        }
        const int r0 = (lane >> 4) * 8;
        v8h pk;
#pragma unroll
        for (int r = 0; r < 8; ++r) pk[r] = (_Float16)(c[r] + biask);
        *(v8h*)&sh_kT[colg][t * 16 + r0] = pk;   // one ds_store_b128
      }
    }
    __syncthreads();

    // (d) scores + mask + softmax (VALU; tiny FLOP share), zero attn accumulator
    for (int i = tid; i < NODES_PER_WG * ND; i += NTHREADS)
      (&sh_attn[0][0])[i] = 0.f;
    for (int i = tid; i < NODES_PER_WG * NH * N_NEI; i += NTHREADS) {
      int b = i / (NH * N_NEI);
      int h = (i / N_NEI) % NH;
      int n = i % N_NEI;
      const int row = b * N_NEI + n;
      const float* qp = &sh_q[b][h * DH];
      float s = 0.f;
#pragma unroll
      for (int d = 0; d < DH; ++d) s += qp[d] * (float)sh_kT[h * DH + d][row];
      sh_w[b][h][n] = s * scale + sh_mb[b][n];
    }
    __syncthreads();
    if (tid < NODES_PER_WG * NH) {
      int b = tid / NH, h = tid % NH;
      float m = -1e30f;
      for (int n = 0; n < N_NEI; ++n) m = fmaxf(m, sh_w[b][h][n]);
      float ssum = 0.f;
      for (int n = 0; n < N_NEI; ++n) {
        float e = __expf(sh_w[b][h][n] - m);
        sh_w[b][h][n] = e; ssum += e;
      }
      float inv = 1.f / ssum;
      for (int n = 0; n < N_NEI; ++n) sh_w[b][h][n] *= inv;
    }
    __syncthreads();

    // (e) V projection fused with weighted reduction: attn += w * (kcomb@Wv+bv)
    {
      const int col0 = wave * 16;
      const int colg = col0 + (lane & 15);
      const int h = colg >> 5;   // head owning this column
      const float biasv = bv[l * HD + colg];
      v16h bfrag[6];
#pragma unroll
      for (int kf = 0; kf < 6; ++kf)
        bfrag[kf] = load_b_frag_t(WvT, CD, 32 * kf, col0, lane);
      for (int t = 0; t < MTILES; ++t) {
        v8f c = {};
#pragma unroll
        for (int kf = 0; kf < 6; ++kf) {
          v16h a = load_a_frag_lds(&sh_kcomb[t * 16][0], CD, 32 * kf, lane);
          c = wmma_f16(a, bfrag[kf], c);
        }
        const int r0 = (lane >> 4) * 8;
#pragma unroll
        for (int r = 0; r < 8; ++r) {
          int flat = t * 16 + r0 + r;
          int b = flat / N_NEI, n = flat % N_NEI;
          atomicAdd(&sh_attn[b][colg], sh_w[b][h][n] * (c[r] + biasv));
        }
      }
    }
    __syncthreads();

    // (f) attn @ Wo + bo, residual + LN1
    for (int i = tid; i < 16 * ND; i += NTHREADS) {
      int r = i >> 7, c = i & 127;
      sh_a16[r][c] = (_Float16)((r < NODES_PER_WG) ? sh_attn[r][c] : 0.f);
    }
    __syncthreads();
    {
      const int col0 = wave * 16;
      const int colg = col0 + (lane & 15);
      v8f c = {};
#pragma unroll
      for (int kf = 0; kf < 4; ++kf) {
        v16h a = load_a_frag_lds(&sh_a16[0][0], CD, 32 * kf, lane);
        v16h bf = load_b_frag_t(WoT, HD, 32 * kf, col0, lane);
        c = wmma_f16(a, bf, c);
      }
      const float bias = bo[l * ND + colg];
      const int r0 = (lane >> 4) * 8;
#pragma unroll
      for (int r = 0; r < 8; ++r) sh_c[r0 + r][colg] = c[r] + bias;
    }
    __syncthreads();
    residual_layernorm(sh_x, sh_c, g1 + l * ND, be1 + l * ND, sh_red, tid);

    // (g) FFN: hidden = relu(x@W1+b1); y = hidden@W2+b2; residual + LN2
    for (int i = tid; i < 16 * ND; i += NTHREADS) {
      int r = i >> 7, c = i & 127;
      sh_a16[r][c] = (_Float16)((r < NODES_PER_WG) ? sh_x[r][c] : 0.f);
    }
    __syncthreads();
    for (int tt = 0; tt < 2; ++tt) {
      const int col0 = (wave + tt * 8) * 16;
      const int colg = col0 + (lane & 15);
      v8f c = {};
#pragma unroll
      for (int kf = 0; kf < 4; ++kf) {
        v16h a = load_a_frag_lds(&sh_a16[0][0], CD, 32 * kf, lane);
        v16h bf = load_b_frag_t(W1T, ND, 32 * kf, col0, lane);
        c = wmma_f16(a, bf, c);
      }
      const float bias = b1[l * 2 * ND + colg];
      const int r0 = (lane >> 4) * 8;
#pragma unroll
      for (int r = 0; r < 8; ++r)
        sh_h16[r0 + r][colg] = (_Float16)fmaxf(c[r] + bias, 0.f);
    }
    __syncthreads();
    {
      const int col0 = wave * 16;
      const int colg = col0 + (lane & 15);
      v8f c = {};
#pragma unroll
      for (int kf = 0; kf < 8; ++kf) {
        v16h a = load_a_frag_lds(&sh_h16[0][0], 2 * ND, 32 * kf, lane);
        v16h bf = load_b_frag_t(W2T, 2 * ND, 32 * kf, col0, lane);
        c = wmma_f16(a, bf, c);
      }
      const float bias = b2[l * ND + colg];
      const int r0 = (lane >> 4) * 8;
#pragma unroll
      for (int r = 0; r < 8; ++r) sh_c[r0 + r][colg] = c[r] + bias;
    }
    __syncthreads();
    residual_layernorm(sh_x, sh_c, g2 + l * ND, be2 + l * ND, sh_red, tid);
  }

  // ---------------- Classifier head ----------------
  for (int i = tid; i < 16 * ND; i += NTHREADS) {
    int r = i >> 7, c = i & 127;
    sh_a16[r][c] = (_Float16)((r < NODES_PER_WG) ? sh_x[r][c] : 0.f);
  }
  __syncthreads();
  if (wave < 4) {   // wave-uniform branch: EXEC all-ones inside
    const int col0 = wave * 16;
    const int colg = col0 + (lane & 15);
    v8f c = {};
#pragma unroll
    for (int kf = 0; kf < 4; ++kf) {
      v16h a = load_a_frag_lds(&sh_a16[0][0], CD, 32 * kf, lane);
      v16h bf = load_b_frag_t(wsh + OFF_WC1, ND, 32 * kf, col0, lane);
      c = wmma_f16(a, bf, c);
    }
    const float bias = bc1[colg];
    const int r0 = (lane >> 4) * 8;
#pragma unroll
    for (int r = 0; r < 8; ++r) sh_c[r0 + r][colg] = fmaxf(c[r] + bias, 0.f);
  }
  __syncthreads();
  if (tid < NODES_PER_WG && (b0 + tid) < B) {
    float s = bc2[0];
#pragma unroll
    for (int j = 0; j < ND / 2; ++j) s += sh_c[tid][j] * Wc2[j];
    out[b0 + tid] = 1.f / (1.f + __expf(-s));
  }
}

extern "C" void kernel_launch(void* const* d_in, const int* in_sizes, int n_in,
                              void* d_out, int out_size, void* d_ws, size_t ws_size,
                              hipStream_t stream) {
  const float* node_features     = (const float*)d_in[0];
  const float* neighbor_features = (const float*)d_in[1];
  const float* node_times        = (const float*)d_in[2];
  const float* neighbor_times    = (const float*)d_in[3];
  const float* time_freq         = (const float*)d_in[4];
  const float* time_phase        = (const float*)d_in[5];
  const float* Wq  = (const float*)d_in[6];
  const float* bq  = (const float*)d_in[7];
  const float* Wk  = (const float*)d_in[8];
  const float* bk  = (const float*)d_in[9];
  const float* Wv  = (const float*)d_in[10];
  const float* bv  = (const float*)d_in[11];
  const float* Wo  = (const float*)d_in[12];
  const float* bo  = (const float*)d_in[13];
  const float* W1  = (const float*)d_in[14];
  const float* b1  = (const float*)d_in[15];
  const float* W2  = (const float*)d_in[16];
  const float* b2  = (const float*)d_in[17];
  const float* g1  = (const float*)d_in[18];
  const float* be1 = (const float*)d_in[19];
  const float* g2  = (const float*)d_in[20];
  const float* be2 = (const float*)d_in[21];
  const float* Wc1 = (const float*)d_in[22];
  const float* bc1 = (const float*)d_in[23];
  const float* Wc2 = (const float*)d_in[24];
  const float* bc2 = (const float*)d_in[25];
  const int*   mask = (const int*)d_in[26];
  float* out = (float*)d_out;
  _Float16* wsh = (_Float16*)d_ws;   // 624 KB of f16 transposed weights

  const int B = in_sizes[0] / ND;
  const int grid = (B + NODES_PER_WG - 1) / NODES_PER_WG;

  prep_weights<<<256, 256, 0, stream>>>(Wq, Wk, Wv, Wo, W1, W2, Wc1, wsh);
  tgat_fused<<<grid, NTHREADS, 0, stream>>>(
      node_features, neighbor_features, node_times, neighbor_times,
      time_freq, time_phase, wsh, bq, bk, bv, bo, b1, b2,
      g1, be1, g2, be2, bc1, Wc2, bc2, mask, out, B);
}
